// PertNet_23759759082205
// MI455X (gfx1250) — matrix operand order repaired and checked by reading, not compiled, not run on hardware
//
#include <hip/hip_runtime.h>
#include <hip/hip_bf16.h>

typedef __attribute__((ext_vector_type(16))) _Float16 v16h;
typedef __attribute__((ext_vector_type(8)))  _Float16 v8h;
typedef __attribute__((ext_vector_type(8)))  float    v8f;

constexpr int NGc = 64;
constexpr int Nc  = 8192;
constexpr int Hc  = 64;
constexpr int Ec  = 262144;
constexpr int Rc  = NGc * Nc;          // 524288 rows
constexpr float LAMc = 0.2f;
constexpr float EPSc = 1e-5f;

// ---- workspace layout (float offsets) ----
constexpr size_t OFF_A    = 0;          // [N,64]  relu(bn_emb)+pert_b
constexpr size_t OFF_PM   = 524288;     // [N,64]  renormed pert_emb
constexpr size_t OFF_PGE  = 1048576;    // [N,64]  SGConv output
constexpr size_t OFF_Z3   = 1572864;    // [R]     final pre-bn3 scalar
constexpr size_t OFF_DINV = 2097152;    // [N]
constexpr size_t OFF_AGG  = 2105344;    // [N,64]  (zeroed)
constexpr size_t OFF_DEG  = 2629632;    // [N]     (zeroed)
constexpr size_t OFF_SV   = 2637824;    // [NG,64] (zeroed)
constexpr size_t OFF_ACC  = 2641920;    // accumulators (zeroed)
constexpr size_t OFF_PRM  = 2646720;    // finalized affine params
constexpr size_t OFF_PACK = 2647488;    // f16 packed B fragments
constexpr int ZLEN = (int)(OFF_PRM - OFF_AGG);  // 541376 floats zeroed

// ACC sub-offsets
constexpr int A_EMBS=0, A_EMBQ=64, A_PBTS=128, A_PBTQ=192, A_PBS=256, A_PBQ=320,
              A_X3S=384, A_GRAM=448, A_Z2S=4544, A_Z2Q=4608, A_Z3S=4672, A_Z3Q=4673;
// PRM sub-offsets: [a|b] pairs of 64 (or 1)
constexpr int P_AB0=0, P_ABPBT=128, P_ABPB=256, P_B1EFF=384, P_AB2=512, P_AB3=640;

// ------------------------------------------------------------------ utils
__global__ void k_zero(float* __restrict__ p, int n) {
  for (int i = blockIdx.x * blockDim.x + threadIdx.x; i < n; i += gridDim.x * blockDim.x)
    p[i] = 0.f;
}

// generic BN finalize: a = g*rsqrt(var+eps), b = beta - mean*a
__global__ void k_fin_bn(const float* __restrict__ S, const float* __restrict__ Q,
                         const float* __restrict__ g, const float* __restrict__ b,
                         float denomInv, float* __restrict__ oa, float* __restrict__ ob, int n) {
  int c = blockIdx.x * blockDim.x + threadIdx.x;
  if (c < n) {
    float m = S[c] * denomInv;
    float v = Q[c] * denomInv - m * m;
    float a = g[c] * rsqrtf(v + EPSc);
    oa[c] = a; ob[c] = b[c] - m * a;
  }
}

// pack static B matrices (W2^T and sg_W^T) into per-lane WMMA fragments
__global__ void k_pack_static(const float* __restrict__ rw2, const float* __restrict__ sgW,
                              _Float16* __restrict__ b2p, _Float16* __restrict__ bsp) {
  int t = blockIdx.x * blockDim.x + threadIdx.x;
  if (t < 8192) {  // B2: K=128 (4 ktiles), N=64 (4 ntiles)
    int f = t >> 9, kt = f >> 2, nt = f & 3;
    int L = (t >> 4) & 31, j = t & 15;
    int k = kt * 32 + (L >> 4) * 16 + j;
    int n = nt * 16 + (L & 15);
    b2p[t] = (_Float16)rw2[n * 128 + k];
  }
  if (t < 4096) {  // Bsg: K=64 (2 ktiles), N=64 (4 ntiles)
    int f = t >> 9, kt = f >> 2, nt = f & 3;
    int L = (t >> 4) & 31, j = t & 15;
    int k = kt * 32 + (L >> 4) * 16 + j;
    int n = nt * 16 + (L & 15);
    bsp[t] = (_Float16)sgW[n * 64 + k];
  }
}

// renorm rows (gene + pert emb); accumulate bn_emb column stats on renormed gene emb
__global__ void k_renorm(const float* __restrict__ ge, const float* __restrict__ pe,
                         float* __restrict__ Pm, float* __restrict__ embS, float* __restrict__ embQ) {
  int wave = threadIdx.x >> 5, lane = threadIdx.x & 31;
  int row = blockIdx.x * 8 + wave;
  if (row >= Nc) return;
  int c0 = lane, c1 = lane + 32;
  float g0 = ge[row * 64 + c0], g1 = ge[row * 64 + c1];
  float p0 = pe[row * 64 + c0], p1 = pe[row * 64 + c1];
  float sg = g0 * g0 + g1 * g1, sp = p0 * p0 + p1 * p1;
  for (int o = 16; o >= 1; o >>= 1) { sg += __shfl_xor(sg, o, 32); sp += __shfl_xor(sp, o, 32); }
  float ng = sqrtf(sg), np = sqrtf(sp);
  float scg = ng > 1.f ? 1.f / (ng + 1e-7f) : 1.f;
  float scp = np > 1.f ? 1.f / (np + 1e-7f) : 1.f;
  Pm[row * 64 + c0] = p0 * scp; Pm[row * 64 + c1] = p1 * scp;
  float v0 = g0 * scg, v1 = g1 * scg;
  atomicAdd(&embS[c0], v0); atomicAdd(&embS[c1], v1);
  atomicAdd(&embQ[c0], v0 * v0); atomicAdd(&embQ[c1], v1 * v1);
}

// A[i,c] = relu(bn_emb(renorm(ge))) + pert_b
__global__ void k_makeA(const float* __restrict__ ge, const float* __restrict__ ab0,
                        const float* __restrict__ pb, float* __restrict__ A) {
  int wave = threadIdx.x >> 5, lane = threadIdx.x & 31;
  int row = blockIdx.x * 8 + wave;
  if (row >= Nc) return;
  int c0 = lane, c1 = lane + 32;
  float g0 = ge[row * 64 + c0], g1 = ge[row * 64 + c1];
  float sg = g0 * g0 + g1 * g1;
  for (int o = 16; o >= 1; o >>= 1) sg += __shfl_xor(sg, o, 32);
  float ng = sqrtf(sg);
  float scg = ng > 1.f ? 1.f / (ng + 1e-7f) : 1.f;
  A[row * 64 + c0] = fmaxf(g0 * scg * ab0[c0] + ab0[64 + c0], 0.f) + pb[c0];
  A[row * 64 + c1] = fmaxf(g1 * scg * ab0[c1] + ab0[64 + c1], 0.f) + pb[c1];
}

// ------------------------------------------------------------------ SGConv
__global__ void k_deg(const int* __restrict__ ei, const float* __restrict__ w, float* __restrict__ deg) {
  int e = blockIdx.x * blockDim.x + threadIdx.x;
  if (e < Ec) atomicAdd(&deg[ei[Ec + e]], w[e]);
}
__global__ void k_dinv(const float* __restrict__ deg, float* __restrict__ dinv) {
  int i = blockIdx.x * blockDim.x + threadIdx.x;
  if (i < Nc) dinv[i] = rsqrtf(deg[i] + 1.0f);   // +1 = self loop weight
}
__global__ void k_scatter(const int* __restrict__ ei, const float* __restrict__ w,
                          const float* __restrict__ dinv, const float* __restrict__ Pm,
                          float* __restrict__ agg) {
  int t = blockIdx.x * blockDim.x + threadIdx.x;
  if (t >= Ec * 4) return;
  int e = t >> 2, q = t & 3;
  int r = ei[e], c = ei[Ec + e];
  float coef = dinv[r] * w[e] * dinv[c];
  #pragma unroll
  for (int j = 0; j < 16; ++j) {
    int cc = q * 16 + j;
    atomicAdd(&agg[c * 64 + cc], coef * Pm[r * 64 + cc]);
  }
}
__global__ void k_self(const float* __restrict__ dinv, const float* __restrict__ Pm,
                       float* __restrict__ agg) {
  int t = blockIdx.x * blockDim.x + threadIdx.x;
  if (t < Nc * 64) { int i = t >> 6; agg[t] += dinv[i] * dinv[i] * Pm[t]; }
}

// A-fragment loader, contiguous K order (lanes 0-15: K 0..15, lanes 16-31: K 16..31)
__device__ inline v16h load_afrag(const _Float16* As, int stride, int lane, int kbase) {
  int r = lane & 15, klo = (lane >> 4) * 16;
  v8h lo = *(const v8h*)&As[r * stride + kbase + klo];
  v8h hi = *(const v8h*)&As[r * stride + kbase + 8 + klo];
  v16h a;
  #pragma unroll
  for (int j = 0; j < 8; ++j) { a[j] = lo[j]; a[8 + j] = hi[j]; }
  return a;
}

// pge = agg @ sg_W^T + sg_b   (WMMA, wave per 16-row tile)
__global__ void k_pge(const float* __restrict__ agg, const _Float16* __restrict__ bsp,
                      const float* __restrict__ sgb, float* __restrict__ pge) {
  __shared__ _Float16 ash[8][1024];
  int wave = threadIdx.x >> 5, lane = threadIdx.x & 31;
  int tile = blockIdx.x * 8 + wave;          // 512 tiles exactly
  _Float16* As = ash[wave];
  int rowbase = tile * 16;
  for (int idx = lane; idx < 1024; idx += 32)
    As[idx] = (_Float16)agg[rowbase * 64 + idx];
  __syncthreads();
  v8f vz = {};
  v8f acc[4] = {vz, vz, vz, vz};
  #pragma unroll
  for (int kt = 0; kt < 2; ++kt) {
    v16h a = load_afrag(As, 64, lane, kt * 32);
    #pragma unroll
    for (int nt = 0; nt < 4; ++nt) {
      v16h b = *(const v16h*)&bsp[(kt * 4 + nt) * 512 + lane * 16];
      acc[nt] = __builtin_amdgcn_wmma_f32_16x16x32_f16(false, a, false, b, (short)0, acc[nt], false, false);
    }
  }
  int mbase = (lane >> 4) * 8, nl = lane & 15;
  #pragma unroll
  for (int nt = 0; nt < 4; ++nt) {
    int c = nt * 16 + nl;
    float bias = sgb[c];
    #pragma unroll
    for (int v = 0; v < 8; ++v)
      pge[(rowbase + mbase + v) * 64 + c] = acc[nt][v] + bias;
  }
}

// s_g = LAM * sum of pge rows of perturbed genes per graph
__global__ void k_svec(const float* __restrict__ x, const float* __restrict__ pge,
                       float* __restrict__ sv) {
  int r = blockIdx.x * blockDim.x + threadIdx.x;
  if (r >= Rc) return;
  float mv = x[2 * r + 1];
  if (mv != 0.f) {
    int g = r >> 13, i = r & (Nc - 1);
    for (int c = 0; c < 64; ++c)
      atomicAdd(&sv[g * 64 + c], LAMc * mv * pge[i * 64 + c]);
  }
}

// per-column sum/sumsq of combined (depth 0) or relu(bn_pbt(combined)) (depth 1)
__global__ void k_rowstats(int depth, const float* __restrict__ A, const float* __restrict__ sv,
                           const float* __restrict__ delta, const float* __restrict__ x,
                           const float* __restrict__ abPBT, float* __restrict__ S, float* __restrict__ Q) {
  __shared__ float red[512];
  int t = threadIdx.x, c = t & 63, rq = t >> 6;
  int rowsPerBlock = Rc / gridDim.x;
  int row0 = blockIdx.x * rowsPerBlock;
  float dc = delta[c], pa = abPBT[c], pbv = abPBT[64 + c];
  float s = 0.f, q = 0.f;
  for (int rr = rq; rr < rowsPerBlock; rr += 4) {
    int r = row0 + rr;
    int i = r & (Nc - 1), g = r >> 13;
    float v = A[i * 64 + c] + sv[g * 64 + c] + x[2 * r + 1] * dc;
    if (depth == 1) v = fmaxf(pa * v + pbv, 0.f);
    s += v; q += v * v;
  }
  red[t] = s; red[256 + t] = q;
  __syncthreads();
  if (rq == 0) {
    atomicAdd(&S[c], red[c] + red[64 + c] + red[128 + c] + red[192 + c]);
    atomicAdd(&Q[c], red[256 + c] + red[320 + c] + red[384 + c] + red[448 + c]);
  }
}

// x3 column sums + 64x64 Gram (enables analytic rbn1 stats)
__global__ void k_x3stats(const float* __restrict__ A, const float* __restrict__ sv,
                          const float* __restrict__ delta, const float* __restrict__ x,
                          const float* __restrict__ abPBT, const float* __restrict__ abPB,
                          float* __restrict__ x3S, float* __restrict__ gram) {
  __shared__ float xr[4][64];
  __shared__ float red[256];
  int t = threadIdx.x, c = t & 63, rq = t >> 6;
  int j0 = t >> 2, k0 = (t & 3) * 16;
  float dc = delta[c], pa = abPBT[c], pbv = abPBT[64 + c], qa = abPB[c], qb = abPB[64 + c];
  float gl[16];
  #pragma unroll
  for (int u = 0; u < 16; ++u) gl[u] = 0.f;
  float cs = 0.f;
  int rowsPerBlock = Rc / gridDim.x;
  int row0 = blockIdx.x * rowsPerBlock;
  for (int rr = 0; rr < rowsPerBlock; rr += 4) {
    int r = row0 + rr + rq;
    int i = r & (Nc - 1), g = r >> 13;
    float v = A[i * 64 + c] + sv[g * 64 + c] + x[2 * r + 1] * dc;
    v = fmaxf(pa * v + pbv, 0.f);
    v = fmaxf(qa * v + qb, 0.f);
    xr[rq][c] = v;
    cs += v;
    __syncthreads();
    #pragma unroll
    for (int rr2 = 0; rr2 < 4; ++rr2) {
      float xj = xr[rr2][j0];
      #pragma unroll
      for (int u = 0; u < 16; ++u) gl[u] += xj * xr[rr2][k0 + u];
    }
    __syncthreads();
  }
  red[t] = cs;
  __syncthreads();
  if (rq == 0) atomicAdd(&x3S[c], red[c] + red[64 + c] + red[128 + c] + red[192 + c]);
  for (int u = 0; u < 16; ++u) atomicAdd(&gram[j0 * 64 + k0 + u], gl[u]);
}

// analytic bn1: mean/var of z1 from mu(x3) + Gram, fold into W1eff + b1eff, pack B1 frags
__global__ void k_finbn1(const float* __restrict__ x3S, const float* __restrict__ gram,
                         const float* __restrict__ rw1, const float* __restrict__ g1,
                         const float* __restrict__ bb1, float* __restrict__ b1eff,
                         _Float16* __restrict__ b1p) {
  __shared__ float mu[64];
  __shared__ float sarr[128];
  int t = threadIdx.x;  // 128
  if (t < 64) mu[t] = x3S[t] / (float)Rc;
  __syncthreads();
  const float* w = &rw1[t * 64];
  float wm = 0.f;
  for (int j = 0; j < 64; ++j) wm += w[j] * mu[j];
  float qq = 0.f;
  for (int j = 0; j < 64; ++j) {
    float wj = w[j]; const float* gr = &gram[j * 64];
    for (int k = 0; k < 64; ++k) qq += wj * w[k] * gr[k];
  }
  float var = qq / (float)Rc - wm * wm;
  float s = g1[t] * rsqrtf(var + EPSc);
  sarr[t] = s;
  b1eff[t] = bb1[t] - wm * s;   // linear bias b1 cancels through BN
  __syncthreads();
  for (int e = t; e < 8192; e += 128) {
    int f = e >> 9, kt = f >> 3, nt = f & 7;
    int L = (e >> 4) & 31, j = e & 15;
    int k = kt * 32 + (L >> 4) * 16 + j;
    int n = nt * 16 + (L & 15);
    b1p[e] = (_Float16)(sarr[n] * rw1[n * 64 + k]);
  }
}

// big fused pass: x3 -> GEMM1(WMMA) -> relu -> GEMM2(WMMA)
// phase 0: accumulate z2 column sum/sumsq; phase 1: h2 = relu(bn2(z2)), z3 = h2.w3+b3, z3 stats
__global__ void k_big(int phase, const float* __restrict__ A, const float* __restrict__ sv,
                      const float* __restrict__ delta, const float* __restrict__ x,
                      const float* __restrict__ abPBT, const float* __restrict__ abPB,
                      const _Float16* __restrict__ b1p, const float* __restrict__ b1eff,
                      const _Float16* __restrict__ b2p, const float* __restrict__ b2lin,
                      const float* __restrict__ ab2, const float* __restrict__ w3,
                      const float* __restrict__ b3,
                      float* __restrict__ z2S, float* __restrict__ z2Q, float* __restrict__ z3,
                      float* __restrict__ z3S, float* __restrict__ z3Q) {
  __shared__ _Float16 ash[8][1024];
  __shared__ _Float16 h1sh[8][2048];
  int wave = threadIdx.x >> 5, lane = threadIdx.x & 31;
  _Float16* As = ash[wave];
  _Float16* Hs = h1sh[wave];
  int nwaves = gridDim.x * 8;
  int wid = blockIdx.x * 8 + wave;
  int mbase = (lane >> 4) * 8, nl = lane & 15;
  int c0 = lane, c1 = lane + 32;

  // warm packed weights into cache hierarchy
  __builtin_prefetch(&b1p[lane * 256], 0, 0);
  __builtin_prefetch(&b2p[lane * 256], 0, 0);

  // per-lane loop invariants (each lane only ever touches fixed columns)
  float d0 = delta[c0], d1 = delta[c1];
  float pa0 = abPBT[c0], pb0 = abPBT[64 + c0], pa1 = abPBT[c1], pb1 = abPBT[64 + c1];
  float qa0 = abPB[c0],  qb0 = abPB[64 + c0],  qa1 = abPB[c1],  qb1 = abPB[64 + c1];
  float be[8];
  #pragma unroll
  for (int nt = 0; nt < 8; ++nt) be[nt] = b1eff[nt * 16 + nl];
  float bl[4], a2c[4], b2c[4], w3c[4];
  #pragma unroll
  for (int nt = 0; nt < 4; ++nt) {
    int c = nt * 16 + nl;
    bl[nt] = b2lin[c]; a2c[nt] = ab2[c]; b2c[nt] = ab2[64 + c]; w3c[nt] = w3[c];
  }
  float bias3 = b3[0];

  float ps[4] = {0, 0, 0, 0}, pq[4] = {0, 0, 0, 0};
  float zs = 0.f, zq = 0.f;
  v8f vz = {};
  for (int tile = wid; tile < Rc / 16; tile += nwaves) {   // uniform trip count
    int rowbase = tile * 16;
    int g = rowbase >> 13;
    int ibase = rowbase & (Nc - 1);
    float sv0 = sv[g * 64 + c0], sv1 = sv[g * 64 + c1];
    float mvown = (lane < 16) ? x[2 * (rowbase + lane) + 1] : 0.f;
    #pragma unroll 4
    for (int m = 0; m < 16; ++m) {
      float mv = __shfl(mvown, m, 32);
      int i = ibase + m;
      float t0 = A[i * 64 + c0] + sv0 + mv * d0;
      float t1 = A[i * 64 + c1] + sv1 + mv * d1;
      t0 = fmaxf(pa0 * t0 + pb0, 0.f);
      t1 = fmaxf(pa1 * t1 + pb1, 0.f);
      t0 = fmaxf(qa0 * t0 + qb0, 0.f);
      t1 = fmaxf(qa1 * t1 + qb1, 0.f);
      As[m * 64 + c0] = (_Float16)t0;
      As[m * 64 + c1] = (_Float16)t1;
    }
    __syncthreads();
    // GEMM1: [16x64] x [64x128]
    v8f acc1[8] = {vz, vz, vz, vz, vz, vz, vz, vz};
    #pragma unroll
    for (int kt = 0; kt < 2; ++kt) {
      v16h a = load_afrag(As, 64, lane, kt * 32);
      #pragma unroll
      for (int nt = 0; nt < 8; ++nt) {
        v16h b = *(const v16h*)&b1p[(kt * 8 + nt) * 512 + lane * 16];
        acc1[nt] = __builtin_amdgcn_wmma_f32_16x16x32_f16(false, a, false, b, (short)0, acc1[nt], false, false);
      }
    }
    #pragma unroll
    for (int nt = 0; nt < 8; ++nt) {
      int c = nt * 16 + nl;
      #pragma unroll
      for (int v = 0; v < 8; ++v)
        Hs[(mbase + v) * 128 + c] = (_Float16)fmaxf(acc1[nt][v] + be[nt], 0.f);
    }
    __syncthreads();
    // GEMM2: [16x128] x [128x64]
    v8f acc2[4] = {vz, vz, vz, vz};
    #pragma unroll
    for (int kt = 0; kt < 4; ++kt) {
      v16h a = load_afrag(Hs, 128, lane, kt * 32);
      #pragma unroll
      for (int nt = 0; nt < 4; ++nt) {
        v16h b = *(const v16h*)&b2p[(kt * 4 + nt) * 512 + lane * 16];
        acc2[nt] = __builtin_amdgcn_wmma_f32_16x16x32_f16(false, a, false, b, (short)0, acc2[nt], false, false);
      }
    }
    if (phase == 0) {
      #pragma unroll
      for (int nt = 0; nt < 4; ++nt) {
        #pragma unroll
        for (int v = 0; v < 8; ++v) { float z = acc2[nt][v] + bl[nt]; ps[nt] += z; pq[nt] += z * z; }
      }
    } else {
      #pragma unroll
      for (int v = 0; v < 8; ++v) {
        float part = 0.f;
        #pragma unroll
        for (int nt = 0; nt < 4; ++nt) {
          float z = acc2[nt][v] + bl[nt];
          float h2 = fmaxf(a2c[nt] * z + b2c[nt], 0.f);
          part += h2 * w3c[nt];
        }
        part += __shfl_xor(part, 1, 16);
        part += __shfl_xor(part, 2, 16);
        part += __shfl_xor(part, 4, 16);
        part += __shfl_xor(part, 8, 16);
        if (nl == 0) {
          float zv = part + bias3;
          z3[rowbase + mbase + v] = zv;
          zs += zv; zq += zv * zv;
        }
      }
    }
    __syncthreads();
  }
  if (phase == 0) {
    for (int nt = 0; nt < 4; ++nt) {
      int c = nt * 16 + nl;
      atomicAdd(&z2S[c], ps[nt]);
      atomicAdd(&z2Q[c], pq[nt]);
    }
  } else if (nl == 0) {
    atomicAdd(z3S, zs); atomicAdd(z3Q, zq);
  }
}

__global__ void k_out(const float* __restrict__ z3, const float* __restrict__ ab3,
                      const float* __restrict__ x, float* __restrict__ out) {
  int r = blockIdx.x * blockDim.x + threadIdx.x;
  if (r < Rc) out[r] = ab3[0] * z3[r] + ab3[1] + x[2 * r];
}

// ------------------------------------------------------------------ launch
extern "C" void kernel_launch(void* const* d_in, const int* in_sizes, int n_in,
                              void* d_out, int out_size, void* d_ws, size_t ws_size,
                              hipStream_t stream) {
  const float* x    = (const float*)d_in[0];
  const int*   ei   = (const int*)  d_in[1];
  const float* ew   = (const float*)d_in[2];
  const float* ge   = (const float*)d_in[3];
  const float* pe   = (const float*)d_in[4];
  const float* pW   = (const float*)d_in[5];
  const float* pb   = (const float*)d_in[6];
  const float* sgW  = (const float*)d_in[7];
  const float* sgb  = (const float*)d_in[8];
  const float* beg  = (const float*)d_in[9];
  const float* beb  = (const float*)d_in[10];
  const float* bptg = (const float*)d_in[11];
  const float* bptb = (const float*)d_in[12];
  const float* bpg  = (const float*)d_in[13];
  const float* bpb2 = (const float*)d_in[14];
  const float* rw1  = (const float*)d_in[15];
  const float* r1g  = (const float*)d_in[17];
  const float* r1b  = (const float*)d_in[18];
  const float* rw2  = (const float*)d_in[19];
  const float* rb2  = (const float*)d_in[20];
  const float* r2g  = (const float*)d_in[21];
  const float* r2b  = (const float*)d_in[22];
  const float* rw3  = (const float*)d_in[23];
  const float* rb3  = (const float*)d_in[24];
  const float* r3g  = (const float*)d_in[25];
  const float* r3b  = (const float*)d_in[26];
  float* out = (float*)d_out;

  float* ws   = (float*)d_ws;
  float* A    = ws + OFF_A;
  float* Pm   = ws + OFF_PM;
  float* pge  = ws + OFF_PGE;
  float* z3   = ws + OFF_Z3;
  float* dinv = ws + OFF_DINV;
  float* agg  = ws + OFF_AGG;
  float* deg  = ws + OFF_DEG;
  float* sv   = ws + OFF_SV;
  float* acc  = ws + OFF_ACC;
  float* prm  = ws + OFF_PRM;
  _Float16* packs = (_Float16*)(ws + OFF_PACK);
  _Float16* b1p = packs;            // 8192 halves
  _Float16* b2p = packs + 8192;     // 8192 halves
  _Float16* bsp = packs + 16384;    // 4096 halves

  const float invN = 1.f / (float)Nc;
  const float invR = 1.f / (float)Rc;

  k_zero<<<1024, 256, 0, stream>>>(agg, ZLEN);
  k_pack_static<<<32, 256, 0, stream>>>(rw2, sgW, b2p, bsp);
  k_renorm<<<Nc / 8, 256, 0, stream>>>(ge, pe, Pm, acc + A_EMBS, acc + A_EMBQ);
  k_fin_bn<<<1, 64, 0, stream>>>(acc + A_EMBS, acc + A_EMBQ, beg, beb, invN,
                                 prm + P_AB0, prm + P_AB0 + 64, 64);
  k_makeA<<<Nc / 8, 256, 0, stream>>>(ge, prm + P_AB0, pb, A);

  k_deg<<<Ec / 256, 256, 0, stream>>>(ei, ew, deg);
  k_dinv<<<Nc / 256, 256, 0, stream>>>(deg, dinv);
  k_scatter<<<Ec * 4 / 256, 256, 0, stream>>>(ei, ew, dinv, Pm, agg);
  k_self<<<Nc * 64 / 256, 256, 0, stream>>>(dinv, Pm, agg);
  k_pge<<<64, 256, 0, stream>>>(agg, bsp, sgb, pge);
  k_svec<<<Rc / 256, 256, 0, stream>>>(x, pge, sv);

  k_rowstats<<<512, 256, 0, stream>>>(0, A, sv, pW, x, prm + P_ABPBT, acc + A_PBTS, acc + A_PBTQ);
  k_fin_bn<<<1, 64, 0, stream>>>(acc + A_PBTS, acc + A_PBTQ, bptg, bptb, invR,
                                 prm + P_ABPBT, prm + P_ABPBT + 64, 64);
  k_rowstats<<<512, 256, 0, stream>>>(1, A, sv, pW, x, prm + P_ABPBT, acc + A_PBS, acc + A_PBQ);
  k_fin_bn<<<1, 64, 0, stream>>>(acc + A_PBS, acc + A_PBQ, bpg, bpb2, invR,
                                 prm + P_ABPB, prm + P_ABPB + 64, 64);
  k_x3stats<<<512, 256, 0, stream>>>(A, sv, pW, x, prm + P_ABPBT, prm + P_ABPB,
                                     acc + A_X3S, acc + A_GRAM);
  k_finbn1<<<1, 128, 0, stream>>>(acc + A_X3S, acc + A_GRAM, rw1, r1g, r1b,
                                  prm + P_B1EFF, b1p);

  k_big<<<512, 256, 0, stream>>>(0, A, sv, pW, x, prm + P_ABPBT, prm + P_ABPB,
                                 b1p, prm + P_B1EFF, b2p, rb2,
                                 prm + P_AB2, rw3, rb3,
                                 acc + A_Z2S, acc + A_Z2Q, z3, acc + A_Z3S, acc + A_Z3Q);
  k_fin_bn<<<1, 64, 0, stream>>>(acc + A_Z2S, acc + A_Z2Q, r2g, r2b, invR,
                                 prm + P_AB2, prm + P_AB2 + 64, 64);
  k_big<<<512, 256, 0, stream>>>(1, A, sv, pW, x, prm + P_ABPBT, prm + P_ABPB,
                                 b1p, prm + P_B1EFF, b2p, rb2,
                                 prm + P_AB2, rw3, rb3,
                                 acc + A_Z2S, acc + A_Z2Q, z3, acc + A_Z3S, acc + A_Z3Q);
  k_fin_bn<<<1, 64, 0, stream>>>(acc + A_Z3S, acc + A_Z3Q, r3g, r3b, invR,
                                 prm + P_AB3, prm + P_AB3 + 1, 1);
  k_out<<<Rc / 256, 256, 0, stream>>>(z3, prm + P_AB3, x, out);
}
